// CachedNNFMLoss_30468497998254
// MI455X (gfx1250) — matrix operand order repaired and checked by reading, not compiled, not run on hardware
//
#include <hip/hip_runtime.h>
#include <hip/hip_bf16.h>

#define CCH 768
#define HW  4096
#define EPSF 1e-8f

typedef __attribute__((ext_vector_type(16))) __bf16 v16bf;
typedef __attribute__((ext_vector_type(8)))  float  v8f;

union FragU { uint4 u[2]; v16bf v; };

__device__ __forceinline__ unsigned short f2bf(float f) {
    // round-to-nearest-even f32 -> bf16
    unsigned u = __float_as_uint(f);
    u += 0x7FFFu + ((u >> 16) & 1u);
    return (unsigned short)(u >> 16);
}

// Kernel 1a: per-pixel reciprocal channel norms (matching part: 1/(sqrt(sum+1e-8)+1e-8)).
// Fully coalesced reads along p.
__global__ void nnfm_norms_kernel(const float* __restrict__ a,
                                  const float* __restrict__ b,
                                  float* __restrict__ rnA,
                                  float* __restrict__ rnB) {
    const int p = blockIdx.x * blockDim.x + threadIdx.x;
    float sa = 0.f, sb = 0.f;
    for (int c = 0; c < CCH; ++c) {
        float av = a[c * HW + p]; sa += av * av;
        float bv = b[c * HW + p]; sb += bv * bv;
    }
    rnA[p] = 1.f / (sqrtf(sa + EPSF) + EPSF);
    rnB[p] = 1.f / (sqrtf(sb + EPSF) + EPSF);
}

// Kernel 1b: scaled transpose + f32->bf16. [c][p] f32 -> [p][c] bf16 via a
// 64x64 LDS tile (65-wide rows to avoid bank conflicts). Both the global
// loads (contiguous p) and stores (contiguous c) are coalesced full lines.
__global__ __launch_bounds__(256) void nnfm_transpose_bf16_kernel(
    const float* __restrict__ a, const float* __restrict__ b,
    const float* __restrict__ rnA, const float* __restrict__ rnB,
    unsigned short* __restrict__ Abf, unsigned short* __restrict__ Bbf) {
    __shared__ float tile[64][65];
    const int which = blockIdx.y;
    const float* src          = which ? b   : a;
    const float* rn           = which ? rnB : rnA;
    unsigned short* dst       = which ? Bbf : Abf;
    const int c0 = (blockIdx.x % (CCH / 64)) * 64;
    const int p0 = (blockIdx.x / (CCH / 64)) * 64;

    #pragma unroll
    for (int k = 0; k < 16; ++k) {
        const int idx = threadIdx.x + k * 256;
        const int i = idx >> 6;        // c offset within tile
        const int j = idx & 63;        // p offset within tile (contiguous -> coalesced)
        tile[i][j] = src[(size_t)(c0 + i) * HW + p0 + j];
    }
    __syncthreads();
    #pragma unroll
    for (int k = 0; k < 16; ++k) {
        const int idx = threadIdx.x + k * 256;
        const int i = idx >> 6;        // p offset within tile
        const int j = idx & 63;        // c offset within tile (contiguous -> coalesced)
        dst[(size_t)(p0 + i) * CCH + c0 + j] = f2bf(tile[j][i] * rn[p0 + i]);
    }
}

// Kernel 2: S = A_n^T B_n via v_wmma_f32_16x16x32_bf16 + fused row-argmax.
// One block per 16-row p tile; 8 waves sweep disjoint 512-wide q ranges.
__global__ __launch_bounds__(256) void nnfm_gemm_argmax_kernel(
    const unsigned short* __restrict__ Abf,
    const unsigned short* __restrict__ Bbf,
    int* __restrict__ zbest,
    int* __restrict__ zbest_out) {
    __shared__ float sVal[8][16];
    __shared__ int   sIdx[8][16];

    const int tid  = threadIdx.x;
    const int wave = tid >> 5;
    const int lane = tid & 31;
    const int half = lane >> 4;   // h: K-half selector per ISA fragment layout
    const int lid  = lane & 15;   // A: row m; B: column n
    const int p0   = blockIdx.x * 16;

    // Preload A fragments: 16 rows x 768 K = 24 fragments (K=32 each).
    // A lane(m,h): VGPR0-3 = K[8h..8h+7], VGPR4-7 = K[16+8h..16+8h+7].
    FragU aF[24];
    const unsigned short* arow = Abf + (size_t)(p0 + lid) * CCH;
    #pragma unroll
    for (int k = 0; k < 24; ++k) {
        const int kb = k * 32;
        aF[k].u[0] = *(const uint4*)(arow + kb + 8 * half);
        aF[k].u[1] = *(const uint4*)(arow + kb + 16 + 8 * half);
    }

    float bestv[8];
    int   besti[8];
    #pragma unroll
    for (int r = 0; r < 8; ++r) { bestv[r] = -3.4e38f; besti[r] = 0; }

    const int q0w = wave * 512;
    for (int qt = 0; qt < 32; ++qt) {
        const int q0 = q0w + qt * 16;
        // B lane(n,h): K[8h..8h+7] then K[16+8h..16+8h+7].
        const unsigned short* brow = Bbf + (size_t)(q0 + lid) * CCH + 8 * half;
        v8f acc = {};
        #pragma unroll
        for (int k = 0; k < 24; ++k) {
            FragU bF;
            bF.u[0] = *(const uint4*)(brow + k * 32);
            bF.u[1] = *(const uint4*)(brow + k * 32 + 16);
            acc = __builtin_amdgcn_wmma_f32_16x16x32_bf16(
                false, aF[k].v, false, bF.v, (short)0, acc, false, false);
        }
        // C/D layout: VGPR r, lanes0-15 -> M=r, lanes16-31 -> M=r+8; N = lane&15.
        const int q = q0 + lid;
        #pragma unroll
        for (int r = 0; r < 8; ++r) {
            if (acc[r] > bestv[r]) { bestv[r] = acc[r]; besti[r] = q; }  // '>' keeps earliest q on ties
        }
    }

    // Reduce over the 16 lanes of each half (the 16 N columns); ties -> smaller q.
    #pragma unroll
    for (int m = 1; m < 16; m <<= 1) {
        #pragma unroll
        for (int r = 0; r < 8; ++r) {
            float ov = __shfl_xor(bestv[r], m, 32);
            int   oi = __shfl_xor(besti[r], m, 32);
            if (ov > bestv[r] || (ov == bestv[r] && oi < besti[r])) {
                bestv[r] = ov; besti[r] = oi;
            }
        }
    }
    if (lid == 0) {
        #pragma unroll
        for (int r = 0; r < 8; ++r) {
            sVal[wave][r + 8 * half] = bestv[r];
            sIdx[wave][r + 8 * half] = besti[r];
        }
    }
    __syncthreads();
    // Combine the 8 waves' disjoint q ranges (ascending q => ties pick lower wave).
    if (tid < 16) {
        float bv = sVal[0][tid]; int bi = sIdx[0][tid];
        for (int w = 1; w < 8; ++w) {
            float ov = sVal[w][tid]; int oi = sIdx[w][tid];
            if (ov > bv || (ov == bv && oi < bi)) { bv = ov; bi = oi; }
        }
        zbest[p0 + tid] = bi;
        zbest_out[p0 + tid] = bi;
    }
}

// Kernel 3: gather z_new[c][p] = d[c][zbest[p]] and per-p cosine-loss partials
// (loss-side norms use sqrt(sum) + 1e-8, per _cos_loss). Deterministic block reduce.
__global__ void nnfm_gather_loss_kernel(const float* __restrict__ a,
                                        const float* __restrict__ d,
                                        const int* __restrict__ zbest,
                                        float* __restrict__ znew,
                                        float* __restrict__ partial) {
    __shared__ float red[256];
    const int p  = blockIdx.x * blockDim.x + threadIdx.x;
    const int zb = zbest[p];
    float saa = 0.f, stt = 0.f, sat = 0.f;
    for (int c = 0; c < CCH; ++c) {
        float av = a[c * HW + p];
        float tv = d[c * HW + zb];
        znew[c * HW + p] = tv;
        saa += av * av; stt += tv * tv; sat += av * tv;
    }
    float cs = sat / ((sqrtf(saa) + EPSF) * (sqrtf(stt) + EPSF));
    red[threadIdx.x] = 1.f - cs;
    __syncthreads();
    for (int s = 128; s > 0; s >>= 1) {
        if (threadIdx.x < s) red[threadIdx.x] += red[threadIdx.x + s];
        __syncthreads();
    }
    if (threadIdx.x == 0) partial[blockIdx.x] = red[0];
}

__global__ void nnfm_finalize_kernel(const float* __restrict__ partial,
                                     float* __restrict__ loss) {
    float s = 0.f;
    for (int i = 0; i < 16; ++i) s += partial[i];
    loss[0] = s / (float)HW;
}

extern "C" void kernel_launch(void* const* d_in, const int* in_sizes, int n_in,
                              void* d_out, int out_size, void* d_ws, size_t ws_size,
                              hipStream_t stream) {
    const float* a = (const float*)d_in[0];
    const float* b = (const float*)d_in[1];
    const float* d = (const float*)d_in[2];

    // Output: [loss (1 f32)] [z_new (768*4096 f32)] [z_best (4096 i32 bits)]
    float* out   = (float*)d_out;
    float* loss  = out;
    float* znew  = out + 1;
    int*   zbout = (int*)(out + 1 + (size_t)CCH * HW);

    // Workspace: Abf (6 MB) | Bbf (6 MB) | zbest (16 KB) | partials (64 B) | rnA/rnB (16 KB each)
    unsigned short* Abf = (unsigned short*)d_ws;
    unsigned short* Bbf = Abf + (size_t)HW * CCH;
    int*   zbest   = (int*)(Bbf + (size_t)HW * CCH);
    float* partial = (float*)(zbest + HW);
    float* rnA     = partial + 16;
    float* rnB     = rnA + HW;

    nnfm_norms_kernel<<<HW / 256, 256, 0, stream>>>(a, b, rnA, rnB);
    dim3 tgrid((CCH / 64) * (HW / 64), 2);
    nnfm_transpose_bf16_kernel<<<tgrid, 256, 0, stream>>>(a, b, rnA, rnB, Abf, Bbf);
    nnfm_gemm_argmax_kernel<<<HW / 16, 256, 0, stream>>>(Abf, Bbf, zbest, zbout);
    nnfm_gather_loss_kernel<<<HW / 256, 256, 0, stream>>>(a, d, zbest, znew, partial);
    nnfm_finalize_kernel<<<1, 1, 0, stream>>>(partial, loss);
}
